// RAPN_48017734369825
// MI455X (gfx1250) — matrix operand order repaired and checked by reading, not compiled, not run on hardware
//
#include <hip/hip_runtime.h>

typedef __bf16 bf16_t;
typedef __attribute__((ext_vector_type(16))) __bf16 v16bf;
typedef __attribute__((ext_vector_type(8)))  float  v8f;

// ---------------- output layout (flat f32, reference return order) ----------
constexpr size_t O0  = 0;       // score_nor      [16,1024]
constexpr size_t O1  = 16384;   // score_abn      [16,1024]
constexpr size_t O2  = 32768;   // nor_label      [16,1024]
constexpr size_t O3  = 49152;   // abn_label      [16,1024]
constexpr size_t O4  = 65536;   // abn_score_h    [16,64]
constexpr size_t O5  = 66560;   // abn_label_hard [16,64]
constexpr size_t O6  = 67584;   // abn_score_cn   [16,64]
constexpr size_t O7  = 68608;   // abn_score_ca   [16,64]
constexpr size_t O8  = 69632;   // nor_feat_cn    [16,64,128]
constexpr size_t O9  = 200704;  // abn_feat_cn    [16,64,128]
constexpr size_t O10 = 331776;  // abn_feat_ca    [16,64,128]

// ---------------------------------------------------------------------------
// Fused GEMM:  C[M,N] = relu( A[M,K] * W[N,K]^T + bias[N] )
// A f32 or bf16; W/bias f32 (converted to bf16 in flight).
// Block 256 thr (8 wave32), block tile 128x128, wave tile 32x64, K-step 32.
// Double-buffered LDS (one barrier per K-step); v_wmma_f32_16x16x32_bf16.
// N_FULL=true removes all N-guards (N multiple of 128).
// ---------------------------------------------------------------------------
template<bool A_BF16, bool OUT_F32, bool OUT_BF16, bool N_FULL>
__global__ __launch_bounds__(256)
void gemm_relu_wmma(const void* __restrict__ Ap, const float* __restrict__ Bw,
                    const float* __restrict__ bias,
                    float* __restrict__ Cf, bf16_t* __restrict__ Cb,
                    int M, int N, int K)
{
    __shared__ bf16_t As[2][128][40];   // +8 pad: 16B-aligned rows, bank spread
    __shared__ bf16_t Bs[2][128][40];   // staged as [N][K]

    const int tid = threadIdx.x;
    const int l   = tid & 31;          // lane in wave32
    const int w   = tid >> 5;          // wave id 0..7
    const int wm  = (w & 3) * 32;      // 4 waves tile M
    const int wn  = (w >> 2) * 64;     // 2 waves tile N
    const int lm  = l & 15;
    const int lhi = l >> 4;            // lane half
    const int ak  = lhi * 8;           // A frag K base (ISA 16-bit A 16x32 layout)
    const int bk  = lhi * 8;           // B frag K base (ISA B 32x16 layout)

    // One loader partition serves both 128x32 tiles: 16 elems/thread.
    const int arow = tid >> 1;
    const int acol = (tid & 1) * 16;
    const int gnB  = blockIdx.y * 128 + arow;

    const size_t aRow = (size_t)(blockIdx.x * 128 + arow) * (size_t)K;
    const size_t bRow = (size_t)gnB * (size_t)K;
    const bool   bOk  = N_FULL || (gnB < N);

    const v8f z = {};
    v8f acc[2][4];
    #pragma unroll
    for (int i = 0; i < 2; ++i)
        #pragma unroll
        for (int j = 0; j < 4; ++j) acc[i][j] = z;

    float4 ar[4];   // staged A (f32: 4 vecs, bf16: 2 vecs)
    float4 brg[4];  // staged B (f32)

    auto loadT = [&](int k0) {
        if (A_BF16) {
            const float4* s4 = (const float4*)((const bf16_t*)Ap + aRow + k0 + acol);
            ar[0] = s4[0]; ar[1] = s4[1];
        } else {
            const float4* s4 = (const float4*)((const float*)Ap + aRow + k0 + acol);
            ar[0] = s4[0]; ar[1] = s4[1]; ar[2] = s4[2]; ar[3] = s4[3];
        }
        if (bOk) {
            const float4* s4 = (const float4*)(Bw + bRow + k0 + acol);
            brg[0] = s4[0]; brg[1] = s4[1]; brg[2] = s4[2]; brg[3] = s4[3];
        } else {
            float4 zf = {0.f, 0.f, 0.f, 0.f};
            brg[0] = zf; brg[1] = zf; brg[2] = zf; brg[3] = zf;
        }
    };
    auto storeT = [&](int buf) {
        if (A_BF16) {
            *(float4*)&As[buf][arow][acol]     = ar[0];
            *(float4*)&As[buf][arow][acol + 8] = ar[1];
        } else {
            alignas(16) bf16_t t[16];
            const float* f = (const float*)ar;
            #pragma unroll
            for (int i = 0; i < 16; ++i) t[i] = (bf16_t)f[i];
            *(float4*)&As[buf][arow][acol]     = *(float4*)&t[0];
            *(float4*)&As[buf][arow][acol + 8] = *(float4*)&t[8];
        }
        {
            alignas(16) bf16_t t[16];
            const float* f = (const float*)brg;
            #pragma unroll
            for (int i = 0; i < 16; ++i) t[i] = (bf16_t)f[i];
            *(float4*)&Bs[buf][arow][acol]     = *(float4*)&t[0];
            *(float4*)&Bs[buf][arow][acol + 8] = *(float4*)&t[8];
        }
    };

    const int NK = K >> 5;
    loadT(0);
    storeT(0);
    __syncthreads();

    for (int kb = 0; kb < NK; ++kb) {
        const int cur = kb & 1;
        if (kb + 1 < NK) loadT((kb + 1) << 5);     // global loads overlap WMMAs
        if (kb + 2 < NK) {                          // L2 prefetch two tiles ahead
            const int kp = (kb + 2) << 5;
            if (A_BF16) __builtin_prefetch((const bf16_t*)Ap + aRow + kp + acol, 0, 0);
            else        __builtin_prefetch((const float*)Ap + aRow + kp + acol, 0, 0);
            if (bOk)    __builtin_prefetch(Bw + bRow + kp + acol, 0, 0);
        }

        union Frag { v16bf v; float4 f[2]; };
        Frag a[2], bf4[4];
        #pragma unroll
        for (int i = 0; i < 2; ++i) {
            a[i].f[0] = *(const float4*)&As[cur][wm + i * 16 + lm][ak];
            a[i].f[1] = *(const float4*)&As[cur][wm + i * 16 + lm][ak + 16];
        }
        #pragma unroll
        for (int j = 0; j < 4; ++j) {
            bf4[j].f[0] = *(const float4*)&Bs[cur][wn + j * 16 + lm][bk];
            bf4[j].f[1] = *(const float4*)&Bs[cur][wn + j * 16 + lm][bk + 16];
        }
        #pragma unroll
        for (int i = 0; i < 2; ++i)
            #pragma unroll
            for (int j = 0; j < 4; ++j)
                acc[i][j] = __builtin_amdgcn_wmma_f32_16x16x32_bf16(
                    false, a[i].v, false, bf4[j].v, (short)0, acc[i][j], false, false);

        if (kb + 1 < NK) storeT((kb + 1) & 1);      // commit next buffer
        __syncthreads();
    }

    // ---- epilogue: bias + relu (C/D layout: M = v + 8*lhi, N = lm) ----
    #pragma unroll
    for (int j = 0; j < 4; ++j) {
        int gn = blockIdx.y * 128 + wn + j * 16 + lm;
        if (!N_FULL && gn >= N) continue;
        float bv = bias[gn];
        #pragma unroll
        for (int i = 0; i < 2; ++i) {
            int gmb = blockIdx.x * 128 + wm + i * 16 + lhi * 8;
            #pragma unroll
            for (int v = 0; v < 8; ++v) {
                float x = fmaxf(acc[i][j][v] + bv, 0.0f);
                size_t off = (size_t)(gmb + v) * (size_t)N + gn;
                if (OUT_F32)  Cf[off] = x;
                if (OUT_BF16) Cb[off] = (bf16_t)x;
            }
        }
    }
}

// ------------- causal 5-tap conv over time (32 ch) + sigmoid ---------------
__global__ __launch_bounds__(256)
void conv_score_kernel(const float* __restrict__ a2, const float* __restrict__ w5,
                       const float* __restrict__ b5, float* __restrict__ score)
{
    int g = blockIdx.x * 256 + threadIdx.x;    // 16*1024 tokens
    int b = g >> 10, t = g & 1023;
    float acc = b5[0];
    #pragma unroll
    for (int k = 0; k < 5; ++k) {
        int tt = t - 4 + k;
        if (tt < 0) continue;
        const float* arow = a2 + (size_t)(b * 1024 + tt) * 32;
        #pragma unroll
        for (int c = 0; c < 32; ++c) acc += arow[c] * w5[c * 5 + k];
    }
    score[g] = 1.0f / (1.0f + expf(-acc));
}

// ------------- exact top-64 per row, lax.top_k order & tie-break -----------
__global__ __launch_bounds__(256)
void topk_kernel(const float* __restrict__ score_nor, const float* __restrict__ score_abn,
                 int* __restrict__ idx)
{
    int sel = blockIdx.x >> 4;
    int b   = blockIdx.x & 15;
    __shared__ float key[1024];
    __shared__ float rv[256];
    __shared__ int   ri[256];
    const float* s = (sel == 0) ? score_nor : score_abn;
    for (int j = threadIdx.x; j < 1024; j += 256) {
        float v = s[b * 1024 + j];
        float k;
        if (sel <= 1)      k = -v;                 // confident-normal picks
        else if (sel == 2) k = v;                  // confident-abnormal
        else               k = -fabsf(v - 0.5f);   // hard/ambiguous
        key[j] = k;
    }
    __syncthreads();
    for (int it = 0; it < 64; ++it) {
        float bv = -1e30f; int bi = 0x7fffffff;
        for (int j = threadIdx.x; j < 1024; j += 256) {
            float v = key[j];
            if (v > bv || (v == bv && j < bi)) { bv = v; bi = j; }
        }
        rv[threadIdx.x] = bv; ri[threadIdx.x] = bi;
        __syncthreads();
        for (int sft = 128; sft > 0; sft >>= 1) {
            if (threadIdx.x < sft) {
                float ov = rv[threadIdx.x + sft]; int oi = ri[threadIdx.x + sft];
                if (ov > rv[threadIdx.x] || (ov == rv[threadIdx.x] && oi < ri[threadIdx.x])) {
                    rv[threadIdx.x] = ov; ri[threadIdx.x] = oi;
                }
            }
            __syncthreads();
        }
        if (threadIdx.x == 0) {
            idx[(sel * 16 + b) * 64 + it] = ri[0];
            key[ri[0]] = -2e30f;           // remove winner
        }
        __syncthreads();
    }
}

// ------------- label init (re-done every call; scatter comes later) --------
__global__ void fill_labels_kernel(float* __restrict__ out)
{
    int g = blockIdx.x * 256 + threadIdx.x;   // 16384
    out[O2 + g] = 0.0f;
    out[O3 + g] = 1.0f;
}

// ------------- gathers: feats + scores at top-k indices --------------------
__global__ __launch_bounds__(128)
void gather_kernel(const float* __restrict__ featn, const float* __restrict__ feata,
                   const float* __restrict__ score_abn, const int* __restrict__ idx,
                   float* __restrict__ out, float* __restrict__ afh)
{
    int r = blockIdx.x, b = r >> 6, d = threadIdx.x;
    int t_nn = idx[r], t_cn = idx[1024 + r], t_ca = idx[2048 + r], t_h = idx[3072 + r];
    out[O8  + (size_t)r * 128 + d] = featn[(size_t)(b * 1024 + t_nn) * 128 + d];
    out[O9  + (size_t)r * 128 + d] = feata[(size_t)(b * 1024 + t_cn) * 128 + d];
    out[O10 + (size_t)r * 128 + d] = feata[(size_t)(b * 1024 + t_ca) * 128 + d];
    afh[(size_t)r * 128 + d]       = feata[(size_t)(b * 1024 + t_h)  * 128 + d];
    if (d == 0) {
        out[O4 + r] = score_abn[b * 1024 + t_h];
        out[O6 + r] = score_abn[b * 1024 + t_cn];
        out[O7 + r] = score_abn[b * 1024 + t_ca];
    }
}

// ------------- q = afh @ Wq ; kproj = keys_feat @ Wk -----------------------
__global__ __launch_bounds__(128)
void proj_kernel(const float* __restrict__ afh, const float* __restrict__ out,
                 const float* __restrict__ Wq, const float* __restrict__ Wk,
                 float* __restrict__ q, float* __restrict__ kp)
{
    __shared__ float f[128];
    int r = blockIdx.x, n = threadIdx.x;
    const float* src; const float* W; float* dst;
    if (r < 1024) {                        // q rows
        src = afh + (size_t)r * 128; W = Wq; dst = q + (size_t)r * 128;
    } else {                               // kproj rows (192 per batch)
        int rk = r - 1024; int b = rk / 192; int k = rk % 192;
        size_t base;
        if (k < 64)       base = O8  + (size_t)(b * 64 + k) * 128;
        else if (k < 128) base = O9  + (size_t)(b * 64 + (k - 64)) * 128;
        else              base = O10 + (size_t)(b * 64 + (k - 128)) * 128;
        src = out + base; W = Wk; dst = kp + (size_t)rk * 128;
    }
    f[n] = src[n];
    __syncthreads();
    float acc = 0.0f;
    #pragma unroll 4
    for (int k2 = 0; k2 < 128; ++k2) acc += f[k2] * W[k2 * 128 + n];
    dst[n] = acc;
}

// ------------- softmax attention + label propagation -----------------------
__global__ __launch_bounds__(256)
void attn_kernel(const float* __restrict__ q, const float* __restrict__ kp,
                 float* __restrict__ newh, float* __restrict__ out)
{
    __shared__ float qv[128];
    __shared__ float red[256];
    __shared__ float red2[256];
    int r = blockIdx.x, b = r >> 6, t = threadIdx.x;
    if (t < 128) qv[t] = q[(size_t)r * 128 + t];
    __syncthreads();
    float logit = -1e30f;
    if (t < 192) {
        const float* kr = kp + (size_t)(b * 192 + t) * 128;
        float acc = 0.0f;
        #pragma unroll 4
        for (int d = 0; d < 128; ++d) acc += qv[d] * kr[d];
        logit = acc * 0.08838834764831845f;   // 1/sqrt(128)
    }
    red[t] = logit;
    __syncthreads();
    for (int s = 128; s > 0; s >>= 1) {
        if (t < s) red[t] = fmaxf(red[t], red[t + s]);
        __syncthreads();
    }
    float m = red[0];
    __syncthreads();
    float e = (t < 192) ? expf(logit - m) : 0.0f;
    float lab = 0.0f;                                  // labels: [0, cn, ca]
    if (t >= 64 && t < 128)       lab = out[O6 + b * 64 + (t - 64)];
    else if (t >= 128 && t < 192) lab = out[O7 + b * 64 + (t - 128)];
    red[t] = e; red2[t] = e * lab;
    __syncthreads();
    for (int s = 128; s > 0; s >>= 1) {
        if (t < s) { red[t] += red[t + s]; red2[t] += red2[t + s]; }
        __syncthreads();
    }
    if (t == 0) {
        float ns = red2[0] / red[0];
        newh[r] = ns;
        out[O5 + r] = (ns > 0.5f) ? 1.0f : 0.0f;       // abn_label_hard
    }
}

// ------------- scatter propagated scores into abn_label --------------------
__global__ void scatter_kernel(const int* __restrict__ idx, const float* __restrict__ newh,
                               float* __restrict__ out)
{
    int r = blockIdx.x * 256 + threadIdx.x;
    if (r >= 1024) return;
    int b = r >> 6;
    int t = idx[3072 + r];                 // idx_h
    out[O3 + (size_t)b * 1024 + t] = newh[r];
}

// ---------------------------------------------------------------------------
extern "C" void kernel_launch(void* const* d_in, const int* in_sizes, int n_in,
                              void* d_out, int out_size, void* d_ws, size_t ws_size,
                              hipStream_t stream)
{
    const float* ref_nor = (const float*)d_in[0];
    const float* ref_abn = (const float*)d_in[1];
    const float* w1  = (const float*)d_in[2];
    const float* b1  = (const float*)d_in[3];
    const float* w2  = (const float*)d_in[4];
    const float* b2  = (const float*)d_in[5];
    const float* wa1 = (const float*)d_in[6];
    const float* ba1 = (const float*)d_in[7];
    const float* wa2 = (const float*)d_in[8];
    const float* ba2 = (const float*)d_in[9];
    const float* w5  = (const float*)d_in[10];
    const float* b5  = (const float*)d_in[11];
    const float* Wq  = (const float*)d_in[12];
    const float* Wk  = (const float*)d_in[13];
    float* out = (float*)d_out;

    char* ws = (char*)d_ws;
    const size_t MB = (size_t)1 << 20;
    bf16_t* h1    = (bf16_t*)(ws);            // 16 MB  [16384,512]  (reused per branch)
    float*  featn = (float*) (ws + 16*MB);    //  8 MB  [16384,128] f32
    float*  feata = (float*) (ws + 24*MB);    //  8 MB
    bf16_t* featb = (bf16_t*)(ws + 32*MB);    //  4 MB  bf16 feat (reused per branch)
    bf16_t* a1b   = (bf16_t*)(ws + 36*MB);    //  2 MB  [16384,64] bf16
    float*  a2f   = (float*) (ws + 38*MB);    //  2 MB  [16384,32] f32
    int*    idx   = (int*)   (ws + 40*MB);    // 16 KB  4 x [16,64]
    float*  afh   = (float*) (ws + 41*MB);    // 512 KB abn_feat_h
    float*  qb    = (float*) (ws + 42*MB);    // 512 KB q
    float*  kpb   = (float*) (ws + 43*MB);    // 1.5 MB kproj
    float*  newh  = (float*) (ws + 45*MB);    //  4 KB  propagated hard scores

    for (int br = 0; br < 2; ++br) {
        const float* x = br ? ref_abn : ref_nor;
        float* feat    = br ? feata : featn;
        float* score   = out + (br ? O1 : O0);
        // layer1: [16384,2048] x [2048,512]   (94% of all FLOPs)
        gemm_relu_wmma<false,false,true, true ><<<dim3(128,4), dim3(256), 0, stream>>>(
            x, w1, b1, nullptr, h1, 16384, 512, 2048);
        // layer2: [16384,512] x [512,128] -> f32 feat + bf16 feat
        gemm_relu_wmma<true, true, true, true ><<<dim3(128,1), dim3(256), 0, stream>>>(
            h1, w2, b2, feat, featb, 16384, 128, 512);
        // attn-branch layer1: [16384,128] x [128,64]
        gemm_relu_wmma<true, false,true, false><<<dim3(128,1), dim3(256), 0, stream>>>(
            featb, wa1, ba1, nullptr, a1b, 16384, 64, 128);
        // attn-branch layer2: [16384,64] x [64,32]
        gemm_relu_wmma<true, true,false, false><<<dim3(128,1), dim3(256), 0, stream>>>(
            a1b, wa2, ba2, a2f, nullptr, 16384, 32, 64);
        // temporal conv + sigmoid -> scores straight into d_out
        conv_score_kernel<<<dim3(64), dim3(256), 0, stream>>>(a2f, w5, b5, score);
    }

    topk_kernel<<<dim3(64),  dim3(256), 0, stream>>>(out + O0, out + O1, idx);
    fill_labels_kernel<<<dim3(64), dim3(256), 0, stream>>>(out);
    gather_kernel<<<dim3(1024), dim3(128), 0, stream>>>(featn, feata, out + O1, idx, out, afh);
    proj_kernel<<<dim3(4096), dim3(128), 0, stream>>>(afh, out, Wq, Wk, qb, kpb);
    attn_kernel<<<dim3(1024), dim3(256), 0, stream>>>(qb, kpb, newh, out);
    scatter_kernel<<<dim3(4), dim3(256), 0, stream>>>(idx, newh, out);

    (void)in_sizes; (void)n_in; (void)out_size; (void)ws_size;
}
